// DCN_Align_26130581029332
// MI455X (gfx1250) — matrix operand order repaired and checked by reading, not compiled, not run on hardware
//
#include <hip/hip_runtime.h>
#include <hip/hip_bf16.h>

typedef unsigned short ushort_t;
typedef __attribute__((ext_vector_type(16))) __bf16 v16bf;
typedef __attribute__((ext_vector_type(8)))  float  v8f;

// round-to-nearest-even fp32 -> bf16
__device__ __forceinline__ ushort_t f2bf(float f) {
    union { float f; unsigned u; } cv; cv.f = f;
    unsigned u = cv.u;
    unsigned r = (u + 0x7FFFu + ((u >> 16) & 1u)) >> 16;
    return (ushort_t)r;
}

union Frag16 { v16bf v; uint4 q[2]; ushort_t u[16]; };

// ---------------------------------------------------------------------------
// Weight prep: fp32 [O][I][3][3] -> bf16 [Opad][I*9] (rows >= O zero-filled).
// K-order = i*9 + (ky*3+kx), identical to the source row-major flattening.
// ---------------------------------------------------------------------------
__global__ void prep_weights_kernel(const float* __restrict__ src,
                                    ushort_t* __restrict__ dst,
                                    int O, int Kt, int total) {
    int i = blockIdx.x * blockDim.x + threadIdx.x;
    if (i >= total) return;
    int o = i / Kt;
    dst[i] = (o < O) ? f2bf(src[i]) : (ushort_t)0;
}

// ---------------------------------------------------------------------------
// 3x3 conv (pad=1) as implicit GEMM with V_WMMA_F32_16X16X32_BF16.
//   block = 256 threads = 8 waves; panel = NP pixels of one output row.
//   Each wave owns one 16-pixel N-tile and BOTH 16-row M-tiles (2 accums):
//   per k-step: 4 ds A-loads + 2 ds B-loads -> 2 WMMAs (B reused).
//   Weights (32 rows of current cblk) staged in LDS once per block.
//   Cin>32 means channel-concat of in0 (ch 0..31) and in1 (ch 32..63).
// ---------------------------------------------------------------------------
__global__ __launch_bounds__(256)
void conv3x3_wmma(const float* __restrict__ in0, const float* __restrict__ in1,
                  const ushort_t* __restrict__ wbf, const float* __restrict__ bias,
                  float* __restrict__ out,
                  int Cout, int Hin, int Win, int Hout, int Wout,
                  int stride, int lrelu, int Ktot, int Kp, int nCblk, int NP) {
    extern __shared__ char smem[];
    ushort_t* wlds  = (ushort_t*)smem;          // [32][Ktot] bf16 weight slice
    ushort_t* panel = wlds + 32 * Ktot;         // [NP][Kp]  bf16 im2col panel

    const int x0   = blockIdx.x * NP;
    const int y    = blockIdx.y;
    const int cblk = blockIdx.z % nCblk;
    const int b    = blockIdx.z / nCblk;

    const int wave = threadIdx.x >> 5;
    const int lane = threadIdx.x & 31;

    // ---- stage weight slice into LDS (16B vectors) ----
    {
        const uint4* src = (const uint4*)(wbf + (size_t)cblk * 32 * Ktot);
        uint4* dst = (uint4*)wlds;
        const int nvec = 32 * Ktot / 8;
        for (int i = threadIdx.x; i < nvec; i += 256) dst[i] = src[i];
    }

    // ---- im2col panel: k-outer (8 waves), pixel-inner (coalesced) ----
    for (int k = wave; k < Ktot; k += 8) {
        int c  = k / 9;
        int t  = k - c * 9;
        int iy = y * stride + (t / 3) - 1;
        int dx = (t % 3) - 1;
        const float* src  = (c < 32) ? in0 : in1;
        const float* srow = src + (((size_t)b * 32 + (c & 31)) * Hin + iy) * Win;
        const bool rowok = (unsigned)iy < (unsigned)Hin;
        for (int n = lane; n < NP; n += 32) {
            int ix = (x0 + n) * stride + dx;
            float v = 0.f;
            if (rowok && (unsigned)ix < (unsigned)Win && (x0 + n) < Wout)
                v = srow[ix];
            panel[n * Kp + k] = f2bf(v);
        }
    }
    __syncthreads();

    // ---- WMMA GEMM: wave nt -> pixel tile, 2 M-tiles per wave ----
    const int nt = wave;
    if (nt * 16 < NP) {
        const int row = lane & 15, half = lane >> 4;
        const ushort_t* w0 = wlds + (size_t)row * Ktot;
        const ushort_t* w1 = wlds + (size_t)(16 + row) * Ktot;
        const ushort_t* br = panel + (size_t)(nt * 16 + row) * Kp;

        v8f acc0 = {0.f, 0.f, 0.f, 0.f, 0.f, 0.f, 0.f, 0.f};
        v8f acc1 = {0.f, 0.f, 0.f, 0.f, 0.f, 0.f, 0.f, 0.f};
        for (int kb = 0; kb < Ktot; kb += 32) {
            Frag16 A0, A1, Bf;
            const uint4* a0 = (const uint4*)(w0 + kb);
            const uint4* a1 = (const uint4*)(w1 + kb);
            const uint4* bp = (const uint4*)(br + kb);
            A0.q[0] = a0[half];  A0.q[1] = a0[2 + half];   // K = kb+h*8.., kb+16+h*8..
            A1.q[0] = a1[half];  A1.q[1] = a1[2 + half];
            Bf.q[0] = bp[half];  Bf.q[1] = bp[2 + half];   // K = kb+h*8.., kb+16+h*8..
            acc0 = __builtin_amdgcn_wmma_f32_16x16x32_bf16(false, A0.v, false, Bf.v,
                                                           (short)0, acc0, false, false);
            acc1 = __builtin_amdgcn_wmma_f32_16x16x32_bf16(false, A1.v, false, Bf.v,
                                                           (short)0, acc1, false, false);
        }

        // ---- epilogue: bias (+ leaky relu) ----
        const int ocb = cblk * 32;
        const int px  = x0 + nt * 16 + row;                 // N = lane & 15
        if (px < Wout) {
#pragma unroll
            for (int r = 0; r < 8; ++r) {
                int oc0 = ocb + r + 8 * half;               // C/D layout: M = r + 8*half
                if (oc0 < Cout) {
                    float v = acc0[r] + bias[oc0];
                    if (lrelu) v = (v >= 0.f) ? v : 0.1f * v;
                    out[(((size_t)b * Cout + oc0) * Hout + y) * Wout + px] = v;
                }
                int oc1 = ocb + 16 + r + 8 * half;
                if (oc1 < Cout) {
                    float v = acc1[r] + bias[oc1];
                    if (lrelu) v = (v >= 0.f) ? v : 0.1f * v;
                    out[(((size_t)b * Cout + oc1) * Hout + y) * Wout + px] = v;
                }
            }
        }
    }
}

// ---------------------------------------------------------------------------
// Bilinear x2 upsample, half-pixel centers, edge clamp (jax.image.resize).
// ---------------------------------------------------------------------------
__global__ void upsample2x_kernel(const float* __restrict__ in,
                                  float* __restrict__ out,
                                  int BC, int Hin, int Win) {
    const int Hout = 2 * Hin, Wout = 2 * Win;
    const long long total = (long long)BC * Hout * Wout;
    for (long long i = (long long)blockIdx.x * blockDim.x + threadIdx.x; i < total;
         i += (long long)gridDim.x * blockDim.x) {
        int xo = (int)(i % Wout);
        int yo = (int)((i / Wout) % Hout);
        int bc = (int)(i / ((long long)Wout * Hout));
        float sy = (yo + 0.5f) * 0.5f - 0.5f;
        float sx = (xo + 0.5f) * 0.5f - 0.5f;
        float fy = floorf(sy), fx = floorf(sx);
        int y0 = (int)fy, x0 = (int)fx;
        float wy = sy - fy, wx = sx - fx;
        int y0c = min(max(y0, 0), Hin - 1);
        int y1c = min(max(y0 + 1, 0), Hin - 1);
        int x0c = min(max(x0, 0), Win - 1);
        int x1c = min(max(x0 + 1, 0), Win - 1);
        const float* p = in + (size_t)bc * Hin * Win;
        float v = (1.f - wy) * ((1.f - wx) * p[y0c * Win + x0c] + wx * p[y0c * Win + x1c])
                +         wy * ((1.f - wx) * p[y1c * Win + x0c] + wx * p[y1c * Win + x1c]);
        out[i] = v;
    }
}

// ---------------------------------------------------------------------------
// Modulated deformable conv (DCNv2): coords+sigmoid(mask) cached in LDS,
// bilinear gather * mask -> bf16 panel, then WMMA GEMM (M=32,K=288,N=NP).
// om channel layout: off_y = g*9+k, off_x = 36+g*9+k, mask = 72+g*9+k.
// GEMM K-order (g*Cg + c)*9 + k  ==  w_dcn's native [O][I][3][3] flattening.
// ---------------------------------------------------------------------------
__global__ __launch_bounds__(256)
void deform_wmma(const float* __restrict__ x, const float* __restrict__ om,
                 const ushort_t* __restrict__ wbf, const float* __restrict__ bias,
                 float* __restrict__ out, int H, int W, int NP) {
    const int Ktot = 288, Kp = 296;
    extern __shared__ char smem[];
    ushort_t* wlds = (ushort_t*)smem;                // [32][288] bf16 weights
    float* pyA = (float*)(wlds + 32 * Ktot);         // [36][NP]
    float* pxA = pyA + 36 * NP;
    float* mkA = pxA + 36 * NP;
    ushort_t* panel = (ushort_t*)(mkA + 36 * NP);    // [NP][Kp] bf16

    const int x0 = blockIdx.x * NP;
    const int yy = blockIdx.y;
    const int b  = blockIdx.z;

    const int wave = threadIdx.x >> 5;
    const int lane = threadIdx.x & 31;

    // ---- stage w_dcn slice into LDS ----
    {
        const uint4* src = (const uint4*)wbf;
        uint4* dst = (uint4*)wlds;
        const int nvec = 32 * Ktot / 8;
        for (int i = threadIdx.x; i < nvec; i += 256) dst[i] = src[i];
    }

    // ---- stage 1: sampling coords + sigmoid(mask), per (g,tap) x pixel ----
    for (int gk = wave; gk < 36; gk += 8) {
        int k = gk % 9;
        float dyf = (float)(k / 3 - 1);
        float dxf = (float)(k % 3 - 1);
        size_t ch  = (size_t)H * W;
        size_t rowoff = (size_t)yy * W;
        const float* py_p = om + ((size_t)b * 108 + gk)      * ch + rowoff;
        const float* px_p = om + ((size_t)b * 108 + 36 + gk) * ch + rowoff;
        const float* mk_p = om + ((size_t)b * 108 + 72 + gk) * ch + rowoff;
        for (int n = lane; n < NP; n += 32) {
            int xx = x0 + n;
            float oy = 0.f, ox = 0.f, mm = 0.f;
            if (xx < W) { oy = py_p[xx]; ox = px_p[xx]; mm = mk_p[xx]; }
            pyA[gk * NP + n] = oy + (float)yy + dyf;
            pxA[gk * NP + n] = ox + (float)xx + dxf;
            mkA[gk * NP + n] = 1.f / (1.f + __expf(-mm));
        }
    }
    __syncthreads();

    // ---- stage 2: bilinear gather * mask -> bf16 panel ----
    for (int kidx = wave; kidx < Ktot; kidx += 8) {
        int cin = kidx / 9;
        int t   = kidx - cin * 9;
        int g   = cin >> 3;                          // Cg = 8
        const float* img = x + ((size_t)b * 32 + cin) * H * W;
        const int base = (g * 9 + t) * NP;
        for (int n = lane; n < NP; n += 32) {
            float py = pyA[base + n], px = pxA[base + n], mk = mkA[base + n];
            float fy = floorf(py), fx = floorf(px);
            int yi = (int)fy, xi = (int)fx;
            float wy = py - fy, wx = px - fx;
            float v = 0.f;
#pragma unroll
            for (int dy = 0; dy < 2; ++dy)
#pragma unroll
                for (int dx = 0; dx < 2; ++dx) {
                    int yc = yi + dy, xc = xi + dx;
                    float wt = (dy ? wy : 1.f - wy) * (dx ? wx : 1.f - wx);
                    if ((unsigned)yc < (unsigned)H && (unsigned)xc < (unsigned)W)
                        v += wt * img[yc * W + xc];
                }
            panel[n * Kp + kidx] = f2bf(v * mk);
        }
    }
    __syncthreads();

    // ---- stage 3: WMMA GEMM (2 M-tiles per wave) ----
    const int nt = wave;
    if (nt * 16 < NP) {
        const int row = lane & 15, half = lane >> 4;
        const ushort_t* w0 = wlds + (size_t)row * Ktot;
        const ushort_t* w1 = wlds + (size_t)(16 + row) * Ktot;
        const ushort_t* br = panel + (size_t)(nt * 16 + row) * Kp;

        v8f acc0 = {0.f, 0.f, 0.f, 0.f, 0.f, 0.f, 0.f, 0.f};
        v8f acc1 = {0.f, 0.f, 0.f, 0.f, 0.f, 0.f, 0.f, 0.f};
        for (int kb = 0; kb < Ktot; kb += 32) {
            Frag16 A0, A1, Bf;
            const uint4* a0 = (const uint4*)(w0 + kb);
            const uint4* a1 = (const uint4*)(w1 + kb);
            const uint4* bp = (const uint4*)(br + kb);
            A0.q[0] = a0[half];  A0.q[1] = a0[2 + half];
            A1.q[0] = a1[half];  A1.q[1] = a1[2 + half];
            Bf.q[0] = bp[half];  Bf.q[1] = bp[2 + half];
            acc0 = __builtin_amdgcn_wmma_f32_16x16x32_bf16(false, A0.v, false, Bf.v,
                                                           (short)0, acc0, false, false);
            acc1 = __builtin_amdgcn_wmma_f32_16x16x32_bf16(false, A1.v, false, Bf.v,
                                                           (short)0, acc1, false, false);
        }

        const int px = x0 + nt * 16 + row;
        if (px < W) {
#pragma unroll
            for (int r = 0; r < 8; ++r) {
                int oc0 = r + 8 * half;
                int oc1 = 16 + r + 8 * half;
                out[(((size_t)b * 32 + oc0) * H + yy) * W + px] = acc0[r] + bias[oc0];
                out[(((size_t)b * 32 + oc1) * H + yy) * W + px] = acc1[r] + bias[oc1];
            }
        }
    }
}

// ---------------------------------------------------------------------------
// Host orchestration
// ---------------------------------------------------------------------------
extern "C" void kernel_launch(void* const* d_in, const int* in_sizes, int n_in,
                              void* d_out, int out_size, void* d_ws, size_t ws_size,
                              hipStream_t stream) {
    (void)in_sizes; (void)n_in; (void)out_size; (void)ws_size;
    const int B = 4, H = 192, W = 192, H2 = 96, W2 = 96, H4 = 48, W4 = 48, NF = 32;

    const float* fea1 = (const float*)d_in[0];
    const float* fea2 = (const float*)d_in[1];
    const float* w1_1 = (const float*)d_in[2];  const float* b1_1 = (const float*)d_in[3];
    const float* w2_1 = (const float*)d_in[4];  const float* b2_1 = (const float*)d_in[5];
    const float* w3_1 = (const float*)d_in[6];  const float* b3_1 = (const float*)d_in[7];
    const float* w4_1 = (const float*)d_in[8];  const float* b4_1 = (const float*)d_in[9];
    const float* w6_1 = (const float*)d_in[10]; const float* b6_1 = (const float*)d_in[11];
    const float* w7_1 = (const float*)d_in[12]; const float* b7_1 = (const float*)d_in[13];
    const float* w1_2 = (const float*)d_in[14]; const float* b1_2 = (const float*)d_in[15];
    const float* w2_2 = (const float*)d_in[16]; const float* b2_2 = (const float*)d_in[17];
    const float* w3_2 = (const float*)d_in[18]; const float* b3_2 = (const float*)d_in[19];
    const float* w4_2 = (const float*)d_in[20]; const float* b4_2 = (const float*)d_in[21];
    const float* w_om = (const float*)d_in[22]; const float* b_om = (const float*)d_in[23];
    const float* w_dcn = (const float*)d_in[24]; const float* b_dcn = (const float*)d_in[25];

    // --- workspace carve-out (aligned 256B) ---
    size_t cur = 0;
    char* ws = (char*)d_ws;
    auto alloc = [&](size_t bytes) -> char* {
        char* p = ws + cur;
        cur += (bytes + 255) & ~(size_t)255;
        return p;
    };
    const size_t SZ192 = (size_t)B * NF * H * W * 4;
    const size_t SZ96  = (size_t)B * NF * H2 * W2 * 4;
    const size_t SZ48  = (size_t)B * NF * H4 * W4 * 4;
    const size_t SZom  = (size_t)B * 108 * H * W * 4;

    float* bufA     = (float*)alloc(SZ192);   // t_off, later reused as up1
    float* bufOff1  = (float*)alloc(SZ192);   // off1, later reused as base_offset
    float* bufOff2a = (float*)alloc(SZ96);
    float* bufOff2  = (float*)alloc(SZ96);
    float* bufOff3a = (float*)alloc(SZ48);
    float* bufOff3  = (float*)alloc(SZ48);
    float* bufUp2   = (float*)alloc(SZ96);
    float* bufT2    = (float*)alloc(SZ96);
    float* bufT3    = (float*)alloc(SZ96);
    float* bufT4    = (float*)alloc(SZ192);
    float* bufOm    = (float*)alloc(SZom);

    ushort_t* wb1_1 = (ushort_t*)alloc((size_t)32 * 576 * 2);
    ushort_t* wb2_1 = (ushort_t*)alloc((size_t)32 * 288 * 2);
    ushort_t* wb3_1 = (ushort_t*)alloc((size_t)32 * 288 * 2);
    ushort_t* wb4_1 = (ushort_t*)alloc((size_t)32 * 288 * 2);
    ushort_t* wb6_1 = (ushort_t*)alloc((size_t)32 * 288 * 2);
    ushort_t* wb7_1 = (ushort_t*)alloc((size_t)32 * 288 * 2);
    ushort_t* wb1_2 = (ushort_t*)alloc((size_t)32 * 576 * 2);
    ushort_t* wb2_2 = (ushort_t*)alloc((size_t)32 * 288 * 2);
    ushort_t* wb3_2 = (ushort_t*)alloc((size_t)32 * 576 * 2);
    ushort_t* wb4_2 = (ushort_t*)alloc((size_t)32 * 288 * 2);
    ushort_t* wb_om = (ushort_t*)alloc((size_t)128 * 288 * 2);  // 108 rows + zero pad
    ushort_t* wb_dcn = (ushort_t*)alloc((size_t)32 * 288 * 2);

    // --- weight bf16 prep ---
    auto prep = [&](const float* src, ushort_t* dst, int O, int Kt, int Opad) {
        int total = Opad * Kt;
        prep_weights_kernel<<<(total + 255) / 256, 256, 0, stream>>>(src, dst, O, Kt, total);
    };
    prep(w1_1, wb1_1, 32, 576, 32);
    prep(w2_1, wb2_1, 32, 288, 32);
    prep(w3_1, wb3_1, 32, 288, 32);
    prep(w4_1, wb4_1, 32, 288, 32);
    prep(w6_1, wb6_1, 32, 288, 32);
    prep(w7_1, wb7_1, 32, 288, 32);
    prep(w1_2, wb1_2, 32, 576, 32);
    prep(w2_2, wb2_2, 32, 288, 32);
    prep(w3_2, wb3_2, 32, 576, 32);
    prep(w4_2, wb4_2, 32, 288, 32);
    prep(w_om, wb_om, 108, 288, 128);
    prep(w_dcn, wb_dcn, 32, 288, 32);

    // --- conv launcher ---
    auto conv = [&](const float* in0, const float* in1, const ushort_t* wbf,
                    const float* bias, float* out, int Cin, int Cout,
                    int Hin, int Win, int stride, int lrelu) {
        int Hout = Hin / stride, Wout = Win / stride;
        int Ktot = Cin * 9, Kp = Ktot + 8;
        int nCblk = (Cout + 31) / 32;
        int NP = (Wout % 96 == 0) ? 96 : ((Wout + 15) & ~15);
        if (NP > 128) NP = 128;
        dim3 grid((Wout + NP - 1) / NP, Hout, B * nCblk);
        size_t shmem = (size_t)32 * Ktot * 2 + (size_t)NP * Kp * 2;
        conv3x3_wmma<<<grid, 256, shmem, stream>>>(in0, in1, wbf, bias, out,
                                                   Cout, Hin, Win, Hout, Wout,
                                                   stride, lrelu, Ktot, Kp, nCblk, NP);
    };
    auto up2 = [&](const float* in, float* out, int Hin, int Win) {
        long long total = (long long)B * NF * 4 * Hin * Win;
        int blocks = (int)((total + 255) / 256);
        upsample2x_kernel<<<blocks, 256, 0, stream>>>(in, out, B * NF, Hin, Win);
    };

    // --- network ---
    conv(fea1, fea2,   wb1_1, b1_1, bufA,     64, 32, H,  W,  1, 1);  // off
    conv(bufA, nullptr, wb2_1, b2_1, bufOff1, 32, 32, H,  W,  1, 1);  // off1
    conv(bufOff1, nullptr, wb3_1, b3_1, bufOff2a, 32, 32, H,  W,  2, 1);
    conv(bufOff2a, nullptr, wb4_1, b4_1, bufOff2,  32, 32, H2, W2, 1, 1);  // off2
    conv(bufOff2, nullptr, wb6_1, b6_1, bufOff3a,  32, 32, H2, W2, 2, 1);
    conv(bufOff3a, nullptr, wb7_1, b7_1, bufOff3,  32, 32, H4, W4, 1, 1);  // off3
    up2(bufOff3, bufUp2, H4, W4);                                          // up -> H/2
    conv(bufUp2, bufOff2, wb1_2, b1_2, bufT2, 64, 32, H2, W2, 1, 1);
    conv(bufT2, nullptr,  wb2_2, b2_2, bufT3, 32, 32, H2, W2, 1, 1);
    up2(bufT3, bufA, H2, W2);                                              // up -> H (reuse bufA)
    conv(bufA, bufOff1, wb3_2, b3_2, bufT4, 64, 32, H, W, 1, 1);
    conv(bufT4, nullptr, wb4_2, b4_2, bufOff1, 32, 32, H, W, 1, 0);        // base_offset (reuse)
    conv(bufOff1, nullptr, wb_om, b_om, bufOm, 32, 108, H, W, 1, 0);       // offset/mask head

    // --- deformable conv -> d_out ---
    {
        const int NP = 96;
        const int Ktot = 288, Kp = 296;
        dim3 grid((W + NP - 1) / NP, H, B);
        size_t shmem = (size_t)32 * Ktot * 2          // weights
                     + (size_t)3 * 36 * NP * 4        // py/px/mask caches
                     + (size_t)NP * Kp * 2;           // bf16 panel
        deform_wmma<<<grid, 256, shmem, stream>>>(fea2, bufOm, wb_dcn, b_dcn,
                                                  (float*)d_out, H, W, NP);
    }
}
